// BatchGAT_69277822485308
// MI455X (gfx1250) — matrix operand order repaired and checked by reading, not compiled, not run on hardware
//
#include <hip/hip_runtime.h>

// ---------------------------------------------------------------------------
// BatchGAT for MI455X (gfx1250, wave32, WMMA).
// f16 A/B fragments, f32 WMMA accumulation; pointwise math in fp32.
// h' kept in f16 (only used as WMMA B operand); adjacency bit-packed.
// ---------------------------------------------------------------------------

typedef __attribute__((ext_vector_type(16))) _Float16 v16h;
typedef __attribute__((ext_vector_type(8)))  _Float16 v8h;
typedef __attribute__((ext_vector_type(8)))  float    v8f;

#define BB 2
#define CC 2
#define NN 1024
#define HH 8
#define FF 64
#define PITCH 40   // f16 LDS pitch: 80B rows keep 32B lane-gathers 16B-aligned

static __device__ __forceinline__ v8f wmma_f16(v16h a, v16h b, v8f c) {
  // (neg_a, A, neg_b, B, c_mod, C, reuse_a, reuse_b)
  return __builtin_amdgcn_wmma_f32_16x16x32_f16(false, a, false, b, (short)0, c,
                                                false, false);
}

// B fragment = two 8-f16 runs (K = lg*8..lg*8+7, 16+lg*8..16+lg*8+7 for column `col`)
static __device__ __forceinline__ v16h load_bfrag(const _Float16* p) {
  const v8h* q = (const v8h*)p;
  v8h lo = q[0], hi = q[2];            // 2x ds_load_b128
  return __builtin_shufflevector(lo, hi, 0, 1, 2, 3, 4, 5, 6, 7,
                                 8, 9, 10, 11, 12, 13, 14, 15);
}

// ===========================================================================
// Kernel 0: pack (adj | self-loop) into bitmask BM[b][n][w], bit j <-> m=w*32+j
// ===========================================================================
__global__ __launch_bounds__(256)
void gat_pack_adj(const int* __restrict__ adj, unsigned* __restrict__ BM)
{
  int i = blockIdx.x * 256 + threadIdx.x;   // over B*N*32 = 65536 words
  int w = i & 31;
  int n = (i >> 5) & (NN - 1);
  int b = i >> 15;
  const int* ap = adj + ((size_t)b * NN + n) * NN + w * 32;
  unsigned bits = 0;
#pragma unroll
  for (int j = 0; j < 32; ++j)
    bits |= (unsigned)(ap[j] != 0) << j;
  if ((n >> 5) == w) bits |= 1u << (n & 31);   // self-loop
  BM[i] = bits;
}

// ===========================================================================
// Kernel 1/4: h' = X · W (per b,c,h slice: [N x KD]*[KD x 64]) -> f16 HPh,
// plus t = tanh(h'), s[n] = t·a_src, d[n] = t·a_dst.
// 256 threads = 8 waves; wave = 16-row x 64-col stripe.
// ===========================================================================
template<int KD>
__global__ __launch_bounds__(256)
void gat_gemm_scores(const float* __restrict__ X,     // [B*C, N, KD]
                     const float* __restrict__ W,     // [C, H, KD, 64]
                     const float* __restrict__ Asrc,  // [C, H, 64]
                     const float* __restrict__ Adst,  // [C, H, 64]
                     _Float16* __restrict__ HPh,      // [B,C,H,N,64] f16
                     float* __restrict__ Sv,          // [B,C,H,N]
                     float* __restrict__ Dv)          // [B,C,H,N]
{
  __shared__ __attribute__((aligned(16))) _Float16 wT[FF][PITCH]; // W^T f16
  __shared__ float tile_s[8][16][FF + 1];   // per-wave h' tile for scores
  __shared__ float avec[FF], dvec[FF];

  const int tiles = NN / 128;
  const int bch  = blockIdx.x / tiles;
  const int tile = blockIdx.x % tiles;
  const int b = bch / (CC * HH);
  const int c = (bch / HH) % CC;
  const int h = bch % HH;
  const int wave = threadIdx.x >> 5;
  const int lane = threadIdx.x & 31;
  const int lm = lane & 15, lg = lane >> 4;
  const int n0 = tile * 128 + wave * 16;

  if (threadIdx.x < FF) {
    avec[threadIdx.x] = Asrc[(c * HH + h) * FF + threadIdx.x];
    dvec[threadIdx.x] = Adst[(c * HH + h) * FF + threadIdx.x];
  }

  v8f acc[4] = {};

  const float* xrow  = X + ((size_t)(b * CC + c) * NN + n0 + lm) * KD;
  const float* wbase = W + (size_t)(c * HH + h) * KD * FF;

  for (int k0 = 0; k0 < KD; k0 += 32) {
    __syncthreads();
    for (int i = threadIdx.x; i < 32 * FF; i += 256) {
      int r = i >> 6, col = i & 63;
      wT[col][r] = (_Float16)wbase[(size_t)(k0 + r) * FF + col];
    }
    __syncthreads();

    // A fragment: two contiguous 8-float runs per lane (ISA 16-bit A layout)
    float4 r0 = *(const float4*)(xrow + k0 + lg * 8);
    float4 r1 = *(const float4*)(xrow + k0 + lg * 8 + 4);
    float4 r2 = *(const float4*)(xrow + k0 + 16 + lg * 8);
    float4 r3 = *(const float4*)(xrow + k0 + 16 + lg * 8 + 4);
    v16h af;
    af[0]  = (_Float16)r0.x; af[1]  = (_Float16)r0.y;
    af[2]  = (_Float16)r0.z; af[3]  = (_Float16)r0.w;
    af[4]  = (_Float16)r1.x; af[5]  = (_Float16)r1.y;
    af[6]  = (_Float16)r1.z; af[7]  = (_Float16)r1.w;
    af[8]  = (_Float16)r2.x; af[9]  = (_Float16)r2.y;
    af[10] = (_Float16)r2.z; af[11] = (_Float16)r2.w;
    af[12] = (_Float16)r3.x; af[13] = (_Float16)r3.y;
    af[14] = (_Float16)r3.z; af[15] = (_Float16)r3.w;

#pragma unroll
    for (int ct = 0; ct < 4; ++ct) {
      v16h bf = load_bfrag(&wT[ct * 16 + lm][lg * 8]);
      acc[ct] = wmma_f16(af, bf, acc[ct]);
    }
  }

  // D layout: vgpr j -> row j + 8*(lane/16), col = lane%16
#pragma unroll
  for (int ct = 0; ct < 4; ++ct)
#pragma unroll
    for (int j = 0; j < 8; ++j) {
      int row = j + 8 * lg;
      int col = ct * 16 + lm;
      float v = acc[ct][j];
      HPh[((size_t)bch * NN + n0 + row) * FF + col] = (_Float16)v;
      tile_s[wave][row][col] = v;
    }
  __syncthreads();

  // scores: lanes<->rows via lane%16; halves split the 64 cols, shfl-combine
  float sa = 0.f, da = 0.f;
  const int ob = lg * 32;
#pragma unroll 8
  for (int o = 0; o < 32; ++o) {
    float t = tanhf(tile_s[wave][lm][ob + o]);
    sa += t * avec[ob + o];
    da += t * dvec[ob + o];
  }
  sa += __shfl_xor(sa, 16);
  da += __shfl_xor(da, 16);
  if (lane < 16) {
    Sv[(size_t)bch * NN + n0 + lm] = sa;
    Dv[(size_t)bch * NN + n0 + lm] = da;
  }
}

// ===========================================================================
// Kernel 2/5: per-row softmax stats (max, sum) over masked leaky-relu scores.
// One wave per (b,c,h,n) row; online max/sum + wave32 shuffle reduction.
// ===========================================================================
__global__ __launch_bounds__(256)
void gat_softmax_stats(const float* __restrict__ Sv, const float* __restrict__ Dv,
                       const unsigned* __restrict__ BM,
                       float* __restrict__ Mx, float* __restrict__ Sm)
{
  const int wave = threadIdx.x >> 5, lane = threadIdx.x & 31;
  const int gidx = blockIdx.x * 8 + wave;   // row over B*C*H*N
  const int bch = gidx >> 10;               // / N
  const int n = gidx & (NN - 1);
  const int b = bch / (CC * HH);
  const float sn = Sv[gidx];
  const float* drow = Dv + (size_t)bch * NN;
  const unsigned* brow = BM + ((size_t)b * NN + n) * 32;

  float mx = -3.0e38f, sum = 0.f;
  for (int it = 0; it < NN / 32; ++it) {
    int m = it * 32 + lane;
    float a = sn + drow[m];
    float l = a >= 0.f ? a : 0.2f * a;
    bool edge = (brow[it] >> lane) & 1u;
    float e = edge ? l : -1.0e12f;
    if (e > mx) { sum *= __expf(mx - e); mx = e; }
    sum += __expf(e - mx);
  }
#pragma unroll
  for (int off = 16; off; off >>= 1) {
    float om = __shfl_xor(mx, off);
    float os = __shfl_xor(sum, off);
    float nm = fmaxf(mx, om);
    sum = sum * __expf(mx - nm) + os * __expf(om - nm);
    mx = nm;
  }
  if (lane == 0) { Mx[gidx] = mx; Sm[gidx] = sum; }
}

// ===========================================================================
// Kernel 3/6: out[n,:] = sum_m softmax(attn)[n,m] * h'[m,:].
// Wave = 32 rows x 64 cols (2 P-fragments, 8 WMMAs/step). P regenerated from
// (s,d,bitmask,stats); B = transposed-f16 h' m-tile in LDS (2x ds_load_b128).
// EPI 0: ELU + head-transpose to x1[B,C,N,512]; EPI 1: out2[B,C,H,N,64].
// ===========================================================================
template<int EPI>
__global__ __launch_bounds__(256)
void gat_aggregate(const _Float16* __restrict__ HPh, const float* __restrict__ Sv,
                   const float* __restrict__ Dv, const float* __restrict__ Mx,
                   const float* __restrict__ Sm, const unsigned* __restrict__ BM,
                   float* __restrict__ Out)
{
  __shared__ __attribute__((aligned(16))) _Float16 hpT[FF][PITCH]; // h'^T f16
  __shared__ float dd_s[32];
  __shared__ unsigned bm_s[256];

  const int tiles = NN / 256;               // 4
  const int bch  = blockIdx.x / tiles;
  const int tile = blockIdx.x % tiles;
  const int b = bch / (CC * HH);
  const int c = (bch / HH) % CC;
  const int h = bch % HH;
  const int wave = threadIdx.x >> 5, lane = threadIdx.x & 31;
  const int lm = lane & 15, lg = lane >> 4;
  const int rowbase = tile * 256;
  const int n0 = rowbase + wave * 32;

  int   row[2];  float sn[2], mxr[2], inv[2];
#pragma unroll
  for (int f = 0; f < 2; ++f) {
    row[f] = n0 + f * 16 + lm;
    sn[f]  = Sv[(size_t)bch * NN + row[f]];
    mxr[f] = Mx[(size_t)bch * NN + row[f]];
    inv[f] = 1.0f / Sm[(size_t)bch * NN + row[f]];  // >= 1 (self-loop)
  }

  v8f acc[2][4] = {};

  for (int m0 = 0; m0 < NN; m0 += 32) {
    __syncthreads();
    for (int i = threadIdx.x; i < 32 * FF; i += 256) {
      int r = i >> 6, col = i & 63;
      hpT[col][r] = HPh[((size_t)bch * NN + m0 + r) * FF + col];
    }
    if (threadIdx.x < 32)
      dd_s[threadIdx.x] = Dv[(size_t)bch * NN + m0 + threadIdx.x];
    bm_s[threadIdx.x] =
        BM[((size_t)b * NN + rowbase + threadIdx.x) * 32 + (m0 >> 5)];
    __syncthreads();

    // P tiles (16x32 f16 A-fragments) for both row groups
    v16h pf[2];
#pragma unroll
    for (int f = 0; f < 2; ++f) {
      const unsigned bits = bm_s[row[f] - rowbase];
#pragma unroll
      for (int j = 0; j < 8; ++j)
#pragma unroll
        for (int e = 0; e < 2; ++e) {
          int k = (j < 4) ? (lg * 8 + 2 * j + e)
                          : (16 + lg * 8 + 2 * (j - 4) + e);
          float a = sn[f] + dd_s[k];
          float l = a >= 0.f ? a : 0.2f * a;
          float p = ((bits >> k) & 1u) ? __expf(l - mxr[f]) * inv[f] : 0.f;
          pf[f][2 * j + e] = (_Float16)p;
        }
    }

#pragma unroll
    for (int ct = 0; ct < 4; ++ct) {
      v16h bf = load_bfrag(&hpT[ct * 16 + lm][lg * 8]);
      acc[0][ct] = wmma_f16(pf[0], bf, acc[0][ct]);
      acc[1][ct] = wmma_f16(pf[1], bf, acc[1][ct]);
    }
  }

#pragma unroll
  for (int f = 0; f < 2; ++f)
#pragma unroll
    for (int ct = 0; ct < 4; ++ct)
#pragma unroll
      for (int j = 0; j < 8; ++j) {
        int ro  = n0 + f * 16 + j + 8 * lg;
        int col = ct * 16 + lm;
        float v = acc[f][ct][j];
        if (EPI == 0) {
          v = v > 0.f ? v : expm1f(v);      // ELU
          Out[((size_t)(b * CC + c) * NN + ro) * (HH * FF) + h * FF + col] = v;
        } else {
          Out[((size_t)bch * NN + ro) * FF + col] = v;
        }
      }
}

// ===========================================================================
// Kernel 7: mean over heads -> [B,C,N,F]
// ===========================================================================
__global__ __launch_bounds__(256)
void gat_head_mean(const float* __restrict__ O2, float* __restrict__ Out)
{
  int i = blockIdx.x * 256 + threadIdx.x;   // over B*C*N*F = 262144
  int f = i & 63;
  int n = (i >> 6) & (NN - 1);
  int bc = i >> 16;
  float s = 0.f;
#pragma unroll
  for (int h = 0; h < HH; ++h)
    s += O2[(((size_t)bc * HH + h) * NN + n) * FF + f];
  Out[i] = s * 0.125f;
}

// ===========================================================================
extern "C" void kernel_launch(void* const* d_in, const int* in_sizes, int n_in,
                              void* d_out, int out_size, void* d_ws, size_t ws_size,
                              hipStream_t stream)
{
  (void)in_sizes; (void)n_in; (void)out_size; (void)ws_size;
  const float* x     = (const float*)d_in[0];
  const int*   adj   = (const int*)d_in[1];
  const float* w1    = (const float*)d_in[2];
  const float* asrc1 = (const float*)d_in[3];
  const float* adst1 = (const float*)d_in[4];
  const float* w2    = (const float*)d_in[5];
  const float* asrc2 = (const float*)d_in[6];
  const float* adst2 = (const float*)d_in[7];
  float* out = (float*)d_out;

  const size_t HPN = (size_t)BB * CC * HH * NN * FF;  // 2,097,152
  const size_t RN  = (size_t)BB * CC * HH * NN;       //   32,768

  // workspace carve-up (~25.25 MiB)
  char* p = (char*)d_ws;
  auto carve = [&](size_t bytes) {
    void* r = (void*)p;
    p += (bytes + 255) & ~(size_t)255;
    return r;
  };
  _Float16* hp1 = (_Float16*)carve(HPN * sizeof(_Float16));
  _Float16* hp2 = (_Float16*)carve(HPN * sizeof(_Float16));
  float*    x1  = (float*)carve(HPN * sizeof(float));   // [B,C,N,512]
  float*    o2  = (float*)carve(HPN * sizeof(float));   // [B,C,H,N,64]
  float*    s1  = (float*)carve(RN * sizeof(float));
  float*    d1  = (float*)carve(RN * sizeof(float));
  float*    mx1 = (float*)carve(RN * sizeof(float));
  float*    sm1 = (float*)carve(RN * sizeof(float));
  float*    s2  = (float*)carve(RN * sizeof(float));
  float*    d2  = (float*)carve(RN * sizeof(float));
  float*    mx2 = (float*)carve(RN * sizeof(float));
  float*    sm2 = (float*)carve(RN * sizeof(float));
  unsigned* bm  = (unsigned*)carve((size_t)BB * NN * 32 * sizeof(unsigned));

  const dim3 blk(256);
  const dim3 pack_grid((BB * NN * 32) / 256);            // 256
  const dim3 gemm_grid(BB * CC * HH * (NN / 128));       // 256
  const dim3 stats_grid(RN / 8);                         // 4096
  const dim3 agg_grid(BB * CC * HH * (NN / 256));        // 128
  const dim3 mean_grid((BB * CC * NN * FF) / 256);       // 1024

  gat_pack_adj<<<pack_grid, blk, 0, stream>>>(adj, bm);

  gat_gemm_scores<FF><<<gemm_grid, blk, 0, stream>>>(x, w1, asrc1, adst1,
                                                     hp1, s1, d1);
  gat_softmax_stats<<<stats_grid, blk, 0, stream>>>(s1, d1, bm, mx1, sm1);
  gat_aggregate<0><<<agg_grid, blk, 0, stream>>>(hp1, s1, d1, mx1, sm1, bm, x1);

  gat_gemm_scores<HH * FF><<<gemm_grid, blk, 0, stream>>>(x1, w2, asrc2, adst2,
                                                          hp2, s2, d2);
  gat_softmax_stats<<<stats_grid, blk, 0, stream>>>(s2, d2, bm, mx2, sm2);
  gat_aggregate<1><<<agg_grid, blk, 0, stream>>>(hp2, s2, d2, mx2, sm2, bm, o2);

  gat_head_mean<<<mean_grid, blk, 0, stream>>>(o2, out);
}
